// Model_42485816492072
// MI455X (gfx1250) — matrix operand order repaired and checked by reading, not compiled, not run on hardware
//
#include <hip/hip_runtime.h>
#include <hip/hip_bf16.h>

typedef __bf16 bf16;
typedef __attribute__((ext_vector_type(16))) __bf16 bf16x16;
typedef __attribute__((ext_vector_type(8)))  __bf16 bf16x8;
typedef __attribute__((ext_vector_type(8)))  float  f32x8;

#define ND     1024
#define LSEQ   512
#define BATCH  8
#define MROWS  4096      // L*B
#define NVOCAB 32000

#define BM 128
#define BN 128
#define BK 32
#define LDT 40           // padded LDS row stride (elements) to avoid bank conflicts

// ---------------- fp32 -> bf16 convert ----------------
__global__ void cvt_bf16_kernel(const float* __restrict__ s, bf16* __restrict__ d, int n) {
    int i = (blockIdx.x * blockDim.x + threadIdx.x) * 4;
    if (i + 3 < n) {
        float4 v = *(const float4*)(s + i);
        d[i]     = (bf16)v.x;
        d[i + 1] = (bf16)v.y;
        d[i + 2] = (bf16)v.z;
        d[i + 3] = (bf16)v.w;
    } else {
        for (; i < n; ++i) d[i] = (bf16)s[i];
    }
}

// ---------------- embedding gather (fp32 + bf16 copies) ----------------
__global__ void embed_kernel(const float* __restrict__ emb, const int* __restrict__ x,
                             float* __restrict__ hf, bf16* __restrict__ hb) {
    int row = blockIdx.x;                 // 0..4095 == t*B + b
    int tok = x[row];
    const float* src = emb + (size_t)tok * ND;
    float* df = hf + (size_t)row * ND;
    bf16*  db = hb + (size_t)row * ND;
    for (int j = threadIdx.x * 4; j < ND; j += blockDim.x * 4) {
        float4 v4 = *(const float4*)(src + j);
        *(float4*)(df + j) = v4;
        db[j]     = (bf16)v4.x;
        db[j + 1] = (bf16)v4.y;
        db[j + 2] = (bf16)v4.z;
        db[j + 3] = (bf16)v4.w;
    }
}

// ---------------- bf16 WMMA GEMM: C[M,N] = A[M,K] @ B[K,N] (+bias) ----------------
// A row-major bf16, B row-major bf16, f32 accumulate.
// Double-buffered LDS pipeline, one barrier per K-step. Templated epilogue.
template <bool HAS_F, bool HAS_B, bool HAS_BIAS>
__global__ __launch_bounds__(256)
void gemm_bf16_kernel(const bf16* __restrict__ A, const bf16* __restrict__ B,
                      float* __restrict__ Cf, bf16* __restrict__ Cb,
                      const float* __restrict__ bias,
                      int M, int N, int K) {
    __shared__ __align__(16) bf16 lA[2][BM * LDT];   // [row][k], row-major
    __shared__ __align__(16) bf16 lB[2][BN * LDT];   // transposed: [col][k]

    const int tid  = threadIdx.x;
    const int lane = tid & 31;
    const int wave = tid >> 5;
    const int hl   = lane >> 4;      // 0/1 half-wave
    const int idx  = lane & 15;
    const int wm   = wave >> 2;      // 0..1 -> 64-row stripe
    const int wn   = wave & 3;       // 0..3 -> 32-col stripe
    const int bm   = blockIdx.y * BM;
    const int bn   = blockIdx.x * BN;

    // cooperative-load indices
    const int arow = tid >> 1;             // 0..127
    const int akh  = (tid & 1) * 16;       // 0 or 16 (elements)
    const int kp   = tid >> 4;             // 0..15 -> k rows 2kp, 2kp+1
    const int ng   = (tid & 15) * 8;       // col group within tile

    const bf16* Ap = A + (size_t)bm * K + (size_t)arow * K + akh;  // advance by BK per step
    const bf16* Bp = B + bn + (size_t)(2 * kp) * N + ng;           // advance by BK*N per step
    const int numK = K / BK;

    f32x8 acc[4][2] = {};

    // ---- prologue: fetch tile 0 and stage into lds[0] ----
    bf16x16 aReg = *(const bf16x16*)(Ap);
    bf16x8  bReg0 = *(const bf16x8*)(Bp);
    bf16x8  bReg1 = *(const bf16x8*)(Bp + N);

    {
        *(bf16x16*)(&lA[0][arow * LDT + akh]) = aReg;
#pragma unroll
        for (int i = 0; i < 8; ++i) {
            union { bf16 h[2]; unsigned u; } pk;
            pk.h[0] = bReg0[i];
            pk.h[1] = bReg1[i];
            *(unsigned*)(&lB[0][(ng + i) * LDT + 2 * kp]) = pk.u;
        }
    }
    __syncthreads();

    int cur = 0;
    for (int kt = 0; kt < numK; ++kt) {
        const bool more = (kt + 1) < numK;
        // ---- prefetch next tile from global into registers ----
        if (more) {
            const bf16* ap = Ap + (size_t)(kt + 1) * BK;
            const bf16* bp = Bp + (size_t)(kt + 1) * BK * N;
            aReg  = *(const bf16x16*)(ap);
            bReg0 = *(const bf16x8*)(bp);
            bReg1 = *(const bf16x8*)(bp + N);
        }

        // ---- compute from lds[cur] ----
        // A frag: lanes 0-15 hold K {0..7,16..23}; lanes 16-31 hold K {8..15,24..31}
        bf16x16 af[4];
#pragma unroll
        for (int m = 0; m < 4; ++m) {
            const bf16* p = &lA[cur][(wm * 64 + m * 16 + idx) * LDT + hl * 8];
            bf16x8 lo = *(const bf16x8*)(p);
            bf16x8 hi = *(const bf16x8*)(p + 16);
#pragma unroll
            for (int e = 0; e < 8; ++e) { af[m][e] = lo[e]; af[m][e + 8] = hi[e]; }
        }
        // B frag: lanes 0-15 hold K {0..7,16..23}; lanes 16-31 hold K {8..15,24..31}
        bf16x16 bfr[2];
#pragma unroll
        for (int n = 0; n < 2; ++n) {
            const bf16* q = &lB[cur][(wn * 32 + n * 16 + idx) * LDT + hl * 8];
            bf16x8 lo = *(const bf16x8*)(q);
            bf16x8 hi = *(const bf16x8*)(q + 16);
#pragma unroll
            for (int e = 0; e < 8; ++e) { bfr[n][e] = lo[e]; bfr[n][e + 8] = hi[e]; }
        }

#pragma unroll
        for (int m = 0; m < 4; ++m)
#pragma unroll
            for (int n = 0; n < 2; ++n)
                acc[m][n] = __builtin_amdgcn_wmma_f32_16x16x32_bf16(
                    false, af[m], false, bfr[n], (short)0, acc[m][n], false, false);

        // ---- stage prefetched tile into the other buffer ----
        if (more) {
            const int nxt = cur ^ 1;
            *(bf16x16*)(&lA[nxt][arow * LDT + akh]) = aReg;
#pragma unroll
            for (int i = 0; i < 8; ++i) {
                union { bf16 h[2]; unsigned u; } pk;
                pk.h[0] = bReg0[i];
                pk.h[1] = bReg1[i];
                *(unsigned*)(&lB[nxt][(ng + i) * LDT + 2 * kp]) = pk.u;
            }
        }
        __syncthreads();
        cur ^= 1;
    }

    // ---- epilogue: D layout = VGPR r -> M = r + hl*8, lane idx -> N ----
#pragma unroll
    for (int m = 0; m < 4; ++m) {
        const int grow = bm + wm * 64 + m * 16 + hl * 8;
#pragma unroll
        for (int n = 0; n < 2; ++n) {
            const int gcol = bn + wn * 32 + n * 16 + idx;
            float bv = 0.0f;
            if constexpr (HAS_BIAS) bv = bias[gcol];
#pragma unroll
            for (int r = 0; r < 8; ++r) {
                const float val = acc[m][n][r] + bv;
                const size_t off = (size_t)(grow + r) * N + gcol;
                if constexpr (HAS_F) Cf[off] = val;
                if constexpr (HAS_B) Cb[off] = (bf16)val;
            }
        }
    }
}

// ---------------- SRU recurrence: one thread per (batch, channel) ----------------
template <bool LAYER0>   // LAYER0: residual packed as U[.., 3d:4d]
__global__ void sru_kernel(const float* __restrict__ U, int ncol,
                           const float* __restrict__ residual,
                           const float* __restrict__ v, const float* __restrict__ b,
                           const float* __restrict__ c0,
                           float* __restrict__ hf, bf16* __restrict__ hb,
                           float* __restrict__ cT) {
    int tid = blockIdx.x * blockDim.x + threadIdx.x;   // 0..8191
    int bi = tid >> 10;
    int ch = tid & (ND - 1);
    const float vf  = v[ch];
    const float vr  = v[ND + ch];
    const float bf_ = b[ch];
    const float br_ = b[ND + ch];
    float c = c0[tid];
    for (int t = 0; t < LSEQ; ++t) {
        size_t base = (size_t)(t * BATCH + bi) * ncol;
        size_t hrow = (size_t)(t * BATCH + bi) * ND;
        float u0 = U[base + ch];
        float u1 = U[base + ND + ch];
        float u2 = U[base + 2 * ND + ch];
        float rin;
        if constexpr (LAYER0) rin = U[base + 3 * ND + ch];
        else                  rin = residual[hrow + ch];
        float f = 1.0f / (1.0f + __expf(-(u1 + vf * c + bf_)));
        c = (c - u0) * f + u0;
        float r = 1.0f / (1.0f + __expf(-(u2 + vr * c + br_)));
        float h = (c - rin) * r + rin;
        hf[hrow + ch] = h;
        hb[hrow + ch] = (bf16)h;
    }
    cT[tid] = c;
}

// ---------------- host orchestration ----------------
extern "C" void kernel_launch(void* const* d_in, const int* in_sizes, int n_in,
                              void* d_out, int out_size, void* d_ws, size_t ws_size,
                              hipStream_t stream) {
    const float* emb  = (const float*)d_in[0];
    const float* W0   = (const float*)d_in[1];
    const float* P1   = (const float*)d_in[2];
    const float* P2   = (const float*)d_in[3];
    const float* vv   = (const float*)d_in[4];
    const float* bb   = (const float*)d_in[5];
    const float* outW = (const float*)d_in[6];
    const float* outb = (const float*)d_in[7];
    const float* hid  = (const float*)d_in[8];
    const int*   x    = (const int*)d_in[9];

    float* out = (float*)d_out;
    float* cs  = out + (size_t)MROWS * NVOCAB;   // [DEPTH, B, ND] appended after logits

    char* ws = (char*)d_ws;
    size_t off = 0;
    auto carve = [&](size_t bytes) -> char* {
        char* p = ws + off;
        off += (bytes + 255) & ~(size_t)255;
        return p;
    };
    bf16*  W0b  = (bf16*) carve((size_t)1024 * 4096 * 2);
    bf16*  P1b  = (bf16*) carve((size_t)3 * 1024 * 512 * 2);
    bf16*  P2b  = (bf16*) carve((size_t)3 * 512 * 3072 * 2);
    bf16*  oWb  = (bf16*) carve((size_t)1024 * 32000 * 2);
    float* hAf  = (float*)carve((size_t)MROWS * ND * 4);
    float* hBf  = (float*)carve((size_t)MROWS * ND * 4);
    bf16*  hAb  = (bf16*) carve((size_t)MROWS * ND * 2);
    bf16*  hBb  = (bf16*) carve((size_t)MROWS * ND * 2);
    float* Uf   = (float*)carve((size_t)MROWS * 4096 * 4);
    bf16*  midb = (bf16*) carve((size_t)MROWS * 512 * 2);

    auto cvt = [&](const float* s, bf16* d, int n) {
        cvt_bf16_kernel<<<dim3((n / 4 + 255) / 256), dim3(256), 0, stream>>>(s, d, n);
    };
    cvt(W0,   W0b, 1024 * 4096);
    cvt(P1,   P1b, 3 * 1024 * 512);
    cvt(P2,   P2b, 3 * 512 * 3072);
    cvt(outW, oWb, 1024 * 32000);

    embed_kernel<<<dim3(MROWS), dim3(256), 0, stream>>>(emb, x, hAf, hAb);

    // Layer 0: U = h @ W0 (4096x4096x1024), residual packed in U's 4th chunk
    gemm_bf16_kernel<true, false, false>
        <<<dim3(4096 / BN, MROWS / BM), dim3(256), 0, stream>>>(
            hAb, W0b, Uf, nullptr, nullptr, MROWS, 4096, 1024);
    sru_kernel<true><<<dim3(32), dim3(256), 0, stream>>>(
        Uf, 4096, nullptr, vv, bb, hid, hBf, hBb, cs);

    float* hcf = hBf; bf16* hcb = hBb;   // current h
    float* hnf = hAf; bf16* hnb = hAb;   // next h

    for (int i = 0; i < 3; ++i) {
        gemm_bf16_kernel<false, true, false>
            <<<dim3(512 / BN, MROWS / BM), dim3(256), 0, stream>>>(
                hcb, P1b + (size_t)i * 1024 * 512, nullptr, midb, nullptr, MROWS, 512, 1024);
        gemm_bf16_kernel<true, false, false>
            <<<dim3(3072 / BN, MROWS / BM), dim3(256), 0, stream>>>(
                midb, P2b + (size_t)i * 512 * 3072, Uf, nullptr, nullptr, MROWS, 3072, 512);
        sru_kernel<false><<<dim3(32), dim3(256), 0, stream>>>(
            Uf, 3072, hcf, vv + (size_t)(i + 1) * 2048, bb + (size_t)(i + 1) * 2048,
            hid + (size_t)(i + 1) * 8192, hnf, hnb, cs + (size_t)(i + 1) * 8192);
        float* tf = hcf; hcf = hnf; hnf = tf;
        bf16*  tb = hcb; hcb = hnb; hnb = tb;
    }

    // Output projection: out = h @ outW + outb (4096 x 32000 x 1024)
    gemm_bf16_kernel<true, false, true>
        <<<dim3(NVOCAB / BN, MROWS / BM), dim3(256), 0, stream>>>(
            hcb, oWb, out, nullptr, outb, MROWS, NVOCAB, 1024);
}
